// DeformableTransformerBlock_23811298689584
// MI455X (gfx1250) — compile-verified, matrix operand order checked by reading, not run on hardware
//
#include <hip/hip_runtime.h>
#include <hip/hip_bf16.h>
#include <cstdint>

// ---- problem dims ----
#define NB    2
#define HH    64
#define WW    64
#define DD    256
#define SS    16
#define NHEAD 4
#define HDIM  64
#define MLPH  1024
#define NTOK  (NB*HH*WW)   // 8192

typedef _Float16 v16h __attribute__((ext_vector_type(16)));
typedef _Float16 v8h  __attribute__((ext_vector_type(8)));
typedef float    v8f  __attribute__((ext_vector_type(8)));

// ---------------- WMMA helpers ----------------
// Fragment layout (CDNA5 ISA 7.12.2, 16-bit A 16x32):
//   lane L: M = L%16, K-base = (L/16)*8; VGPR0..3 = halves [base..base+7],
//   VGPR4..7 = halves [base+16..base+23].  B (32x16) mirrors with N in lanes.
// So a fragment = two contiguous 16-byte (8-half) runs at p and p+16.
__device__ __forceinline__ v16h load_frag16(const _Float16* p) {
    v8h lo = *(const v8h*)p;
    v8h hi = *(const v8h*)(p + 16);
    v16h r;
#pragma unroll
    for (int i = 0; i < 8; ++i) { r[i] = lo[i]; r[i + 8] = hi[i]; }
    return r;
}

__device__ __forceinline__ v8f wmma_f16f32(v16h a, v16h b, v8f c) {
    return __builtin_amdgcn_wmma_f32_16x16x32_f16(
        /*neg_a=*/false, a, /*neg_b=*/false, b,
        /*c_mod=*/(short)0, c, /*reuse_a=*/false, /*reuse_b=*/false);
}

__device__ __forceinline__ float gelu_tanh(float x) {
    float x3 = x * x * x;
    return 0.5f * x * (1.0f + tanhf(0.7978845608028654f * (x + 0.044715f * x3)));
}

// ---------------- weight prep: fp32 [rows][cols] -> f16 [cols][rows] ----------------
__global__ __launch_bounds__(256) void transpose_to_f16(const float* __restrict__ src,
                                                        _Float16* __restrict__ dst,
                                                        int rows, int cols) {
    int i = blockIdx.x * 256 + threadIdx.x;
    if (i >= rows * cols) return;
    int r = i / cols, c = i - r * cols;
    dst[(size_t)c * rows + r] = (_Float16)src[i];
}

// ---------------- stage 0: offsets + bilinear sampling -> kv (f16), hs (f16) ----------------
__global__ __launch_bounds__(256) void sample_kernel(
    const float* __restrict__ hs, const float* __restrict__ ehs,
    const float* __restrict__ W_off, const float* __restrict__ b_off,
    const float* __restrict__ W_kvp, const float* __restrict__ b_kvp,
    _Float16* __restrict__ kv_h, _Float16* __restrict__ hs_h)
{
    __shared__ float hsl[DD];
    __shared__ float offl[SS * 2];
    __shared__ int   y0l[SS], x0l[SS], y1l[SS], x1l[SS];
    __shared__ float wyl[SS], wxl[SS];

    const int t = blockIdx.x;
    const int tid = threadIdx.x;
    const int n = t >> 12;               // / (H*W)
    const int rem = t & 4095;
    const int h = rem >> 6;
    const int w = rem & 63;

    float hv = hs[(size_t)t * DD + tid];
    hsl[tid] = hv;
    hs_h[(size_t)t * DD + tid] = (_Float16)hv;
    __syncthreads();

    if (tid < SS * 2) {
        float acc = b_off[tid];
#pragma unroll 4
        for (int d = 0; d < DD; ++d) acc += hsl[d] * W_off[d * (SS * 2) + tid];
        // 2*MAX_OFF*sigmoid - MAX_OFF
        offl[tid] = 60.0f / (1.0f + expf(-acc)) - 30.0f;
    }
    __syncthreads();
    if (tid < SS) {
        float y = fminf(fmaxf((float)h + offl[2 * tid + 0], 0.0f), 63.0f);
        float x = fminf(fmaxf((float)w + offl[2 * tid + 1], 0.0f), 63.0f);
        float y0 = floorf(y), x0 = floorf(x);
        int y0i = (int)y0, x0i = (int)x0;
        y0l[tid] = y0i; x0l[tid] = x0i;
        y1l[tid] = min(y0i + 1, 63); x1l[tid] = min(x0i + 1, 63);
        wyl[tid] = y - y0; wxl[tid] = x - x0;
    }
    __syncthreads();

    const int d = tid;
    const float* base = ehs + (size_t)n * HH * WW * DD;
    const float wk0 = W_kvp[d], wk1 = W_kvp[DD + d], bk = b_kvp[d];
#pragma unroll 4
    for (int s = 0; s < SS; ++s) {
        int y0i = y0l[s], x0i = x0l[s], y1i = y1l[s], x1i = x1l[s];
        float wy = wyl[s], wx = wxl[s];
        float v00 = base[((size_t)(y0i * WW + x0i)) * DD + d];
        float v01 = base[((size_t)(y0i * WW + x1i)) * DD + d];
        float v10 = base[((size_t)(y1i * WW + x0i)) * DD + d];
        float v11 = base[((size_t)(y1i * WW + x1i)) * DD + d];
        float bl = (1.0f - wy) * (1.0f - wx) * v00 + (1.0f - wy) * wx * v01 +
                   wy * (1.0f - wx) * v10 + wy * wx * v11;
        float kvv = bl + offl[2 * s + 0] * wk0 + offl[2 * s + 1] * wk1 + bk;
        kv_h[((size_t)t * SS + s) * DD + d] = (_Float16)kvv;
    }
}

// ---------------- generic wave-tiled WMMA GEMM ----------------
// C[M x Nc] = A[M x K](f16 row-major) * BT[Nc x K](f16 row-major, i.e. B^T) + bias
// Each wave computes one 16(M) x 64(N) tile. 256 threads = 8 waves per block.
// EP: 0 = store fp32, 2 = tanh-GELU then store f16.
template <int EP>
__global__ __launch_bounds__(256) void gemm_kernel(
    const _Float16* __restrict__ A, const _Float16* __restrict__ BT,
    const float* __restrict__ bias, float* __restrict__ Cf,
    _Float16* __restrict__ Ch, int M, int Nc, int K)
{
    const int wave = blockIdx.x * 8 + (threadIdx.x >> 5);
    const int lane = threadIdx.x & 31;
    const int tilesN = Nc >> 6;
    const int tm = wave / tilesN;
    const int tn = wave - tm * tilesN;
    if (tm * 16 >= M) return;

    const int lm = lane & 15;
    const int kb = (lane >> 4) << 3;   // 0 or 8 (also the D-fragment row offset)
    const int row = tm * 16 + lm;

    v8f z = {0.f,0.f,0.f,0.f,0.f,0.f,0.f,0.f};
    v8f acc[4] = {z, z, z, z};

    const _Float16* Ap = A + (size_t)row * K + kb;
    const _Float16* Bp[4];
#pragma unroll
    for (int tnn = 0; tnn < 4; ++tnn)
        Bp[tnn] = BT + (size_t)(tn * 64 + tnn * 16 + lm) * K + kb;

    for (int kk = 0; kk < K; kk += 32) {
        v16h a = load_frag16(Ap + kk);
#pragma unroll
        for (int tnn = 0; tnn < 4; ++tnn) {
            v16h b = load_frag16(Bp[tnn] + kk);
            acc[tnn] = wmma_f16f32(a, b, acc[tnn]);
        }
    }

#pragma unroll
    for (int tnn = 0; tnn < 4; ++tnn) {
        int c = tn * 64 + tnn * 16 + lm;
        float bs = bias[c];
#pragma unroll
        for (int r = 0; r < 8; ++r) {
            int m = tm * 16 + kb + r;
            float val = acc[tnn][r] + bs;
            if (EP == 0) {
                Cf[(size_t)m * Nc + c] = val;
            } else {
                Ch[(size_t)m * Nc + c] = (_Float16)gelu_tanh(val);
            }
        }
    }
}

// ---------------- fused per-token K/V projection + attention ----------------
// One block (8 waves) per token: K = kv*Wk+bk, V = kv*Wv+bv via WMMA into LDS,
// then scores/softmax/P*V in VALU. Emits attn (pre-output-proj) as f16.
#define KV_LD 264   // f16 row stride (528B: 16B aligned, rotates banks)
#define F_LD  260   // f32 row stride for K/V tiles in LDS
__global__ __launch_bounds__(256) void attn_fused_kernel(
    const _Float16* __restrict__ kv, const float* __restrict__ q,
    const _Float16* __restrict__ WkT, const _Float16* __restrict__ WvT,
    const float* __restrict__ bk, const float* __restrict__ bv,
    _Float16* __restrict__ attn_out)
{
    __shared__ __align__(16) _Float16 kvl[SS * KV_LD];
    __shared__ float ql[DD];
    __shared__ float Kl[SS * F_LD];
    __shared__ float Vl[SS * F_LD];
    __shared__ float sc[NHEAD * SS];
    __shared__ float pl[NHEAD * SS];

    const int t = blockIdx.x;
    const int tid = threadIdx.x;

    // stage kv tile (16 x 256 f16) into padded LDS
    const uint32_t* src = (const uint32_t*)(kv + (size_t)t * SS * DD);
    for (int i = tid; i < SS * (DD / 2); i += 256) {
        int r = i >> 7, c2 = i & 127;
        ((uint32_t*)&kvl[r * KV_LD])[c2] = src[r * (DD / 2) + c2];
    }
    ql[tid] = q[(size_t)t * DD + tid];
    __syncthreads();

    const int wave = tid >> 5;
    const int lane = tid & 31;
    const int lm = lane & 15;
    const int kb = (lane >> 4) << 3;

#pragma unroll
    for (int i = 0; i < 2; ++i) {
        const int nt = wave * 2 + i;
        const int c = nt * 16 + lm;
        v8f ak = {0.f,0.f,0.f,0.f,0.f,0.f,0.f,0.f};
        v8f av = ak;
        const _Float16* bkp = WkT + (size_t)c * DD + kb;
        const _Float16* bvp = WvT + (size_t)c * DD + kb;
        for (int kk = 0; kk < DD; kk += 32) {
            v16h a  = load_frag16(&kvl[lm * KV_LD + kk + kb]);
            v16h fk = load_frag16(bkp + kk);
            v16h fv = load_frag16(bvp + kk);
            ak = wmma_f16f32(a, fk, ak);
            av = wmma_f16f32(a, fv, av);
        }
        float bK = bk[c], bV = bv[c];
#pragma unroll
        for (int r = 0; r < 8; ++r) {
            Kl[(kb + r) * F_LD + c] = ak[r] + bK;
            Vl[(kb + r) * F_LD + c] = av[r] + bV;
        }
    }
    __syncthreads();

    // scores[a][s] = (q_a . k_{s,a}) / sqrt(HD)
    if (tid < NHEAD * SS) {
        int a = tid >> 4, s = tid & 15;
        float acc = 0.f;
#pragma unroll 8
        for (int e = 0; e < HDIM; ++e)
            acc += ql[a * HDIM + e] * Kl[s * F_LD + a * HDIM + e];
        sc[tid] = acc * 0.125f;
    }
    __syncthreads();
    if (tid < NHEAD * SS) {
        int a = tid >> 4;
        float m = -1e30f;
#pragma unroll
        for (int s2 = 0; s2 < SS; ++s2) m = fmaxf(m, sc[a * SS + s2]);
        float sum = 0.f;
#pragma unroll
        for (int s2 = 0; s2 < SS; ++s2) sum += expf(sc[a * SS + s2] - m);
        pl[tid] = expf(sc[tid] - m) / sum;
    }
    __syncthreads();

    {   // attn[d] = sum_s p[a][s] * V[s][d],  a = d / HD
        int a = tid >> 6;
        float acc = 0.f;
#pragma unroll
        for (int s = 0; s < SS; ++s) acc += pl[a * SS + s] * Vl[s * F_LD + tid];
        attn_out[(size_t)t * DD + tid] = (_Float16)acc;
    }
}

// ---------------- residual + LayerNorm ----------------
__device__ __forceinline__ float block_sum256(float v, float* red) {
#pragma unroll
    for (int o = 16; o > 0; o >>= 1) v += __shfl_xor(v, o, 32);
    __syncthreads();
    if ((threadIdx.x & 31) == 0) red[threadIdx.x >> 5] = v;
    __syncthreads();
    float s = 0.f;
#pragma unroll
    for (int i = 0; i < 8; ++i) s += red[i];
    return s;
}

__global__ __launch_bounds__(256) void add_ln_kernel(
    const float* __restrict__ A, const float* __restrict__ B,
    const float* __restrict__ scale, const float* __restrict__ bias,
    float* __restrict__ outf, _Float16* __restrict__ outh)
{
    __shared__ float red[8];
    const size_t idx = (size_t)blockIdx.x * DD + threadIdx.x;
    float v = A[idx] + B[idx];
    float mean = block_sum256(v, red) * (1.0f / DD);
    float dv = v - mean;
    float var = block_sum256(dv * dv, red) * (1.0f / DD);
    float y = dv * rsqrtf(var + 1e-6f) * scale[threadIdx.x] + bias[threadIdx.x];
    outf[idx] = y;
    if (outh) outh[idx] = (_Float16)y;
}

// ---------------- host launch ----------------
extern "C" void kernel_launch(void* const* d_in, const int* in_sizes, int n_in,
                              void* d_out, int out_size, void* d_ws, size_t ws_size,
                              hipStream_t stream) {
    (void)in_sizes; (void)n_in; (void)out_size; (void)ws_size;
    const float* hs    = (const float*)d_in[0];
    const float* ehs   = (const float*)d_in[1];
    const float* W_off = (const float*)d_in[2];
    const float* b_off = (const float*)d_in[3];
    const float* W_kvp = (const float*)d_in[4];
    const float* b_kvp = (const float*)d_in[5];
    const float* Wq    = (const float*)d_in[6];
    const float* bq    = (const float*)d_in[7];
    const float* Wk    = (const float*)d_in[8];
    const float* bk    = (const float*)d_in[9];
    const float* Wv    = (const float*)d_in[10];
    const float* bv    = (const float*)d_in[11];
    const float* Wo    = (const float*)d_in[12];
    const float* bo    = (const float*)d_in[13];
    const float* ln1s  = (const float*)d_in[14];
    const float* ln1b  = (const float*)d_in[15];
    const float* ln2s  = (const float*)d_in[16];
    const float* ln2b  = (const float*)d_in[17];
    const float* W1    = (const float*)d_in[18];
    const float* b1    = (const float*)d_in[19];
    const float* W2    = (const float*)d_in[20];
    const float* b2    = (const float*)d_in[21];
    float* out = (float*)d_out;

    char* ws = (char*)d_ws;
    size_t off = 0;
    auto alloc = [&](size_t bytes) -> char* {
        char* p = ws + off;
        off = (off + bytes + 255) & ~(size_t)255;
        return p;
    };
    _Float16* hs_h = (_Float16*)alloc((size_t)NTOK * DD * 2);
    _Float16* kv_h = (_Float16*)alloc((size_t)NTOK * SS * DD * 2);
    float*    q_f  = (float*)   alloc((size_t)NTOK * DD * 4);
    _Float16* at_h = (_Float16*)alloc((size_t)NTOK * DD * 2);
    float*    o_f  = (float*)   alloc((size_t)NTOK * DD * 4);
    float*    x_f  = (float*)   alloc((size_t)NTOK * DD * 4);
    _Float16* x_h  = (_Float16*)alloc((size_t)NTOK * DD * 2);
    _Float16* h_h  = (_Float16*)alloc((size_t)NTOK * MLPH * 2);
    float*    m_f  = (float*)   alloc((size_t)NTOK * DD * 4);
    _Float16* WqT  = (_Float16*)alloc((size_t)DD * DD * 2);
    _Float16* WkT  = (_Float16*)alloc((size_t)DD * DD * 2);
    _Float16* WvT  = (_Float16*)alloc((size_t)DD * DD * 2);
    _Float16* WoT  = (_Float16*)alloc((size_t)DD * DD * 2);
    _Float16* W1T  = (_Float16*)alloc((size_t)DD * MLPH * 2);
    _Float16* W2T  = (_Float16*)alloc((size_t)MLPH * DD * 2);

    // weight prep (f32 -> f16 transposed)
    transpose_to_f16<<<(DD * DD + 255) / 256, 256, 0, stream>>>(Wq, WqT, DD, DD);
    transpose_to_f16<<<(DD * DD + 255) / 256, 256, 0, stream>>>(Wk, WkT, DD, DD);
    transpose_to_f16<<<(DD * DD + 255) / 256, 256, 0, stream>>>(Wv, WvT, DD, DD);
    transpose_to_f16<<<(DD * DD + 255) / 256, 256, 0, stream>>>(Wo, WoT, DD, DD);
    transpose_to_f16<<<(DD * MLPH + 255) / 256, 256, 0, stream>>>(W1, W1T, DD, MLPH);
    transpose_to_f16<<<(MLPH * DD + 255) / 256, 256, 0, stream>>>(W2, W2T, MLPH, DD);

    // stage 0: offsets + bilinear sampling
    sample_kernel<<<NTOK, 256, 0, stream>>>(hs, ehs, W_off, b_off, W_kvp, b_kvp, kv_h, hs_h);

    // Q projection: [8192x256] = hs * Wq + bq (fp32 out)
    {
        int blocks = ((NTOK / 16) * (DD / 64) + 7) / 8;
        gemm_kernel<0><<<blocks, 256, 0, stream>>>(hs_h, WqT, bq, q_f, nullptr, NTOK, DD, DD);
    }
    // fused K/V projection + attention
    attn_fused_kernel<<<NTOK, 256, 0, stream>>>(kv_h, q_f, WkT, WvT, bk, bv, at_h);
    // output projection
    {
        int blocks = ((NTOK / 16) * (DD / 64) + 7) / 8;
        gemm_kernel<0><<<blocks, 256, 0, stream>>>(at_h, WoT, bo, o_f, nullptr, NTOK, DD, DD);
    }
    // x = LN1(hs + attn)
    add_ln_kernel<<<NTOK, 256, 0, stream>>>(hs, o_f, ln1s, ln1b, x_f, x_h);
    // MLP1 + GELU (f16 out)
    {
        int blocks = ((NTOK / 16) * (MLPH / 64) + 7) / 8;
        gemm_kernel<2><<<blocks, 256, 0, stream>>>(x_h, W1T, b1, nullptr, h_h, NTOK, MLPH, DD);
    }
    // MLP2 (fp32 out)
    {
        int blocks = ((NTOK / 16) * (DD / 64) + 7) / 8;
        gemm_kernel<0><<<blocks, 256, 0, stream>>>(h_h, W2T, b2, m_f, nullptr, NTOK, DD, MLPH);
    }
    // out = LN2(x + mlp)
    add_ln_kernel<<<NTOK, 256, 0, stream>>>(x_f, m_f, ln2s, ln2b, out, nullptr);
}
